// SymmetricBranchAttention_64785286693352
// MI455X (gfx1250) — matrix operand order repaired and checked by reading, not compiled, not run on hardware
//
#include <hip/hip_runtime.h>
#include <hip/hip_bf16.h>

// ---------------------------------------------------------------------------
// SymmetricBranchAttention on MI455X (gfx1250, wave32, WMMA + TDM).
// GEMMs use v_wmma_f32_16x16x32_f16. Weight operands for the big GEMMs are
// staged into LDS by the Tensor Data Mover (tensor_load_to_lds, double
// buffered, s_wait_tensorcnt) with TDM padding for conflict-free LDS reads.
// Attention-map streaming uses global_prefetch. 192MB L2 carries the rest.
// ---------------------------------------------------------------------------

typedef _Float16 h16;
typedef __attribute__((ext_vector_type(16))) _Float16 v16h;
typedef __attribute__((ext_vector_type(8)))  _Float16 v8h;   // 16B = one b128
typedef __attribute__((ext_vector_type(8)))  float    v8f;
typedef __attribute__((ext_vector_type(4)))  unsigned int u32x4;
typedef __attribute__((ext_vector_type(8)))  int i32x8;
typedef __attribute__((ext_vector_type(4)))  int i32x4;

#define BB 8
#define NN 1024
#define CC 768
#define HH 12
#define DD 64
#define BH (BB*HH)          // 96
#define MTOT (BB*NN)        // 8192
#define SCALE 0.125f        // D^-0.5
#define EPS 1e-8f

// LDS row stride for TDM-staged weight chunks: 32 data halves (64B) + TDM pad
// of 4 dwords (16B) every 16 dwords => 80B = 40 halves. 16B aligned, and
// column stride of 20 dwords gives 16 distinct banks across a 16-lane group.
#define WROW 40

// ---- WMMA fragment loaders (layouts per CDNA5 ISA §7.12.2) -----
__device__ __forceinline__ v16h frag_a(const h16* __restrict__ A, long lda,
                                       int row0, int k0) {
  const int l = threadIdx.x & 31;
  const int r = l & 15, lh = l >> 4;
  const h16* p = A + (long)(row0 + r) * lda + k0 + lh * 16;
  v16h f;
  ((v8h*)&f)[0] = *(const v8h*)p;        // global_load_b128
  ((v8h*)&f)[1] = *(const v8h*)(p + 8);  // global_load_b128
  return f;
}

// B fragment from K-contiguous global storage Bc[n][k]
__device__ __forceinline__ v16h frag_b(const h16* __restrict__ Bc, long ldb,
                                       int col0, int k0) {
  const int l = threadIdx.x & 31;
  const int c = l & 15, lh = l >> 4;
  const h16* p = Bc + (long)(col0 + c) * ldb + k0 + lh * 16;
  v16h f;
  ((v8h*)&f)[0] = *(const v8h*)p;
  ((v8h*)&f)[1] = *(const v8h*)(p + 8);
  return f;
}

// B fragment from a TDM-staged LDS chunk (one 32-k slice, padded rows)
__device__ __forceinline__ v16h frag_b_lds(const h16* wb, int col_local) {
  const int l = threadIdx.x & 31;
  const int c = l & 15, lh = l >> 4;
  const h16* p = wb + (col_local + c) * WROW + lh * 16;
  v16h f;
  ((v8h*)&f)[0] = *(const v8h*)p;        // ds_load_b128
  ((v8h*)&f)[1] = *(const v8h*)(p + 8);  // ds_load_b128
  return f;
}

__device__ __forceinline__ v8f wmma32(v16h a, v16h b, v8f c) {
  return __builtin_amdgcn_wmma_f32_16x16x32_f16(false, a, false, b,
                                                (short)0, c, false, false);
}

// ---- Tensor Data Mover: load one [128 cols x 32 k] f16 weight chunk -------
// D# per CDNA5 ISA: 2D tile, data_size=2B,
// tile_dim0=32 (k, contiguous), tile_dim1=128 (cols), tensor_dim0_stride=768,
// pad_enable, pad_interval=16 dwords, pad_amount=4 dwords.
__device__ __forceinline__ void tdm_load_wchunk(const h16* gsrc,
                                                unsigned lds_off) {
  unsigned long ga = (unsigned long)gsrc;
  u32x4 g0;
  g0[0] = 1u;                                   // count=1, user descriptor
  g0[1] = lds_off;                              // lds_addr (bytes)
  g0[2] = (unsigned)ga;                         // global_addr[31:0]
  g0[3] = (unsigned)((ga >> 32) & 0x01FFFFFFu)  // global_addr[56:32]
        | (2u << 30);                           // type=2 ("image")
  i32x8 g1;
  g1[0] = (1 << 16)                             // data_size = 2B
        | (1 << 20)                             // pad_enable
        | (3 << 22)                             // pad_interval: 16 dwords
        | (3 << 25);                            // pad_amount: 4 dwords
  g1[1] = (768 << 16);                          // tensor_dim0 = 768 (lo16)
  g1[2] = (128 << 16);                          // tensor_dim1 = 128 (lo16)
  g1[3] = (32 << 16);                           // tile_dim0 = 32
  g1[4] = 128;                                  // tile_dim1 = 128, tile_dim2=0
  g1[5] = 768;                                  // tensor_dim0_stride (lo32)
  g1[6] = 0;
  g1[7] = 0;
  i32x4 z4 = {0, 0, 0, 0};
#if __clang_major__ >= 23
  i32x8 z8 = {0, 0, 0, 0, 0, 0, 0, 0};
  __builtin_amdgcn_tensor_load_to_lds(g0, g1, z4, z4, z8, 0);
#else
  __builtin_amdgcn_tensor_load_to_lds(g0, g1, z4, z4, 0);
#endif
}

__device__ __forceinline__ unsigned lds_offset_of(const void* p) {
  return (unsigned)(unsigned long)p;  // low 32 bits of flat LDS address
}

// ---- converts -------------------------------------------------------------
__global__ __launch_bounds__(256) void cvt_f16_kernel(
    const float* __restrict__ src, h16* __restrict__ dst, int n) {
  int i = blockIdx.x * 256 + threadIdx.x;
  if (i < n) dst[i] = (h16)src[i];
}

// dst[n*K + k] = src[k*Ncols + n]  (transpose + f16 convert, for weights)
__global__ __launch_bounds__(256) void cvt_t_kernel(
    const float* __restrict__ src, h16* __restrict__ dst, int K, int Ncols) {
  long i = (long)blockIdx.x * 256 + threadIdx.x;
  if (i < (long)K * Ncols) {
    int n = (int)(i / K), k = (int)(i % K);
    dst[i] = (h16)src[(long)k * Ncols + n];
  }
}

// ---- QKV GEMM: [8192x768]f16 @ [768x2304] -> scatter Q, K, V^T ------------
// block 256 thr = 8 waves (4m x 2n); block tile 128x128; B chunks TDM-staged.
__global__ __launch_bounds__(256) void qkv_kernel(
    const h16* __restrict__ xh,
    const h16* __restrict__ wt0, const h16* __restrict__ wt1,
    h16* __restrict__ Q0, h16* __restrict__ K0, h16* __restrict__ Vt0,
    h16* __restrict__ Q1, h16* __restrict__ K1, h16* __restrict__ Vt1) {
  __shared__ h16 wbuf[2][128 * WROW];
  const int br = blockIdx.z;
  const h16* wt = br ? wt1 : wt0;
  h16* Qm = br ? Q1 : Q0;
  h16* Km = br ? K1 : K0;
  h16* Vt = br ? Vt1 : Vt0;
  const int wave = threadIdx.x >> 5;
  const int row0 = blockIdx.x * 128 + (wave >> 1) * 32;
  const int colB = blockIdx.y * 128;          // block's first output column
  const int colL = (wave & 1) * 64;           // wave's LDS-local column base
  const h16* wslab = wt + (long)colB * CC;    // 128 K-contiguous weight rows

  if (wave == 0) tdm_load_wchunk(wslab, lds_offset_of(&wbuf[0][0]));

  v8f acc[2][4] = {};
  for (int kc = 0; kc < CC / 32; ++kc) {
    const int k0 = kc * 32;
    __builtin_amdgcn_s_wait_tensorcnt((short)0);
    __syncthreads();                          // chunk kc visible to all waves
    if (wave == 0 && kc + 1 < CC / 32)
      tdm_load_wchunk(wslab + (kc + 1) * 32,
                      lds_offset_of(&wbuf[(kc + 1) & 1][0]));
    const h16* wb = &wbuf[kc & 1][0];
    v16h a0 = frag_a(xh, CC, row0, k0);
    v16h a1 = frag_a(xh, CC, row0 + 16, k0);
#pragma unroll
    for (int j = 0; j < 4; ++j) {
      v16h b = frag_b_lds(wb, colL + j * 16);
      acc[0][j] = wmma32(a0, b, acc[0][j]);
      acc[1][j] = wmma32(a1, b, acc[1][j]);
    }
  }
  const int l = threadIdx.x & 31, cl = l & 15, lh = l >> 4;
#pragma unroll
  for (int mi = 0; mi < 2; ++mi)
#pragma unroll
    for (int j = 0; j < 4; ++j)
#pragma unroll
      for (int v = 0; v < 8; ++v) {
        int row = row0 + mi * 16 + lh * 8 + v;
        int col = colB + colL + j * 16 + cl;
        int b  = row >> 10, n = row & (NN - 1);
        int t  = col / CC, rem = col - t * CC;
        int h  = rem >> 6, d = rem & 63;
        long bh = (long)(b * HH + h);
        h16 val = (h16)acc[mi][j][v];
        if (t == 0)      Qm[(bh * NN + n) * DD + d] = val;        // [B,H,N,D]
        else if (t == 1) Km[(bh * NN + n) * DD + d] = val;        // [B,H,N,D]
        else             Vt[(bh * DD + d) * NN + n] = val;        // [B,H,D,N]
      }
}

// ---- scores + fused softmax ----------------------------------------------
// grid (N/32, BH, 2). Workgroup owns 32 full rows x 1024 cols of one (b,h).
__global__ __launch_bounds__(256) void scores_kernel(
    const h16* __restrict__ Q0, const h16* __restrict__ K0,
    const h16* __restrict__ Q1, const h16* __restrict__ K1,
    h16* __restrict__ attn0, h16* __restrict__ attn1) {
  const int br = blockIdx.z;
  const long bh = blockIdx.y;
  const h16* Qb = (br ? Q1 : Q0) + bh * NN * DD;
  const h16* Kb = (br ? K1 : K0) + bh * NN * DD;
  h16* Ab = (br ? attn1 : attn0) + bh * (long)NN * NN;
  const int wave = threadIdx.x >> 5;
  const int row0 = blockIdx.x * 32;
  const int col0 = wave * 128;
  const int l = threadIdx.x & 31, cl = l & 15, lh = l >> 4;

  v8f acc[2][8] = {};
#pragma unroll
  for (int k0 = 0; k0 < DD; k0 += 32) {
    v16h a0 = frag_a(Qb, DD, row0, k0);
    v16h a1 = frag_a(Qb, DD, row0 + 16, k0);
#pragma unroll
    for (int j = 0; j < 8; ++j) {
      v16h b = frag_b(Kb, DD, col0 + j * 16, k0);
      acc[0][j] = wmma32(a0, b, acc[0][j]);
      acc[1][j] = wmma32(a1, b, acc[1][j]);
    }
  }

  __shared__ float redmax[8 * 32];
  __shared__ float redsum[8 * 32];
  float rm[2][8], rs[2][8];

  // scale + clip, per-row max over this wave's 128 cols
#pragma unroll
  for (int mi = 0; mi < 2; ++mi)
#pragma unroll
    for (int v = 0; v < 8; ++v) {
      float m = -1e30f;
#pragma unroll
      for (int j = 0; j < 8; ++j) {
        float s = fminf(fmaxf(acc[mi][j][v] * SCALE, -20.f), 20.f);
        acc[mi][j][v] = s;
        m = fmaxf(m, s);
      }
#pragma unroll
      for (int off = 1; off < 16; off <<= 1)
        m = fmaxf(m, __shfl_xor(m, off, 32));
      rm[mi][v] = m;
    }
  if (cl == 0) {
#pragma unroll
    for (int mi = 0; mi < 2; ++mi)
#pragma unroll
      for (int v = 0; v < 8; ++v)
        redmax[wave * 32 + mi * 16 + lh * 8 + v] = rm[mi][v];
  }
  __syncthreads();
#pragma unroll
  for (int mi = 0; mi < 2; ++mi)
#pragma unroll
    for (int v = 0; v < 8; ++v) {
      float m = -1e30f;
#pragma unroll
      for (int w2 = 0; w2 < 8; ++w2)
        m = fmaxf(m, redmax[w2 * 32 + mi * 16 + lh * 8 + v]);
      rm[mi][v] = m;
    }
  // exp + per-row sum
#pragma unroll
  for (int mi = 0; mi < 2; ++mi)
#pragma unroll
    for (int v = 0; v < 8; ++v) {
      float s = 0.f;
#pragma unroll
      for (int j = 0; j < 8; ++j) {
        float e = __expf(acc[mi][j][v] - rm[mi][v]);
        acc[mi][j][v] = e;
        s += e;
      }
#pragma unroll
      for (int off = 1; off < 16; off <<= 1)
        s += __shfl_xor(s, off, 32);
      rs[mi][v] = s;
    }
  if (cl == 0) {
#pragma unroll
    for (int mi = 0; mi < 2; ++mi)
#pragma unroll
      for (int v = 0; v < 8; ++v)
        redsum[wave * 32 + mi * 16 + lh * 8 + v] = rs[mi][v];
  }
  __syncthreads();
#pragma unroll
  for (int mi = 0; mi < 2; ++mi)
#pragma unroll
    for (int v = 0; v < 8; ++v) {
      float s = 0.f;
#pragma unroll
      for (int w2 = 0; w2 < 8; ++w2)
        s += redsum[w2 * 32 + mi * 16 + lh * 8 + v];
      rs[mi][v] = 1.f / s;
    }
  // normalized probabilities -> f16 attention map
#pragma unroll
  for (int mi = 0; mi < 2; ++mi)
#pragma unroll
    for (int j = 0; j < 8; ++j)
#pragma unroll
      for (int v = 0; v < 8; ++v) {
        long row = row0 + mi * 16 + lh * 8 + v;
        Ab[row * NN + col0 + j * 16 + cl] = (h16)(acc[mi][j][v] * rs[mi][v]);
      }
}

// ---- overlap -> per-row scale (same for both branches) --------------------
// mask = sigmoid(-overlap*temp); softmax rows sum to 1, so the renorm is the
// per-row scalar mask/(mask+EPS), folded into the attn@V epilogue.
__global__ __launch_bounds__(256) void overlap_kernel(
    const h16* __restrict__ attn0, const h16* __restrict__ attn1,
    const float* __restrict__ temp_ptr, float* __restrict__ rowscale) {
  const long bh = blockIdx.y;
  const long row = blockIdx.x;
  const h16* p0 = attn0 + (bh * NN + row) * NN;
  const h16* p1 = attn1 + (bh * NN + row) * NN;
  float s = 0.f;
  for (int i = threadIdx.x; i < NN; i += 256)
    s += (float)p0[i] * (float)p1[i];
  __shared__ float sm[256];
  sm[threadIdx.x] = s;
  __syncthreads();
  for (int st = 128; st > 0; st >>= 1) {
    if (threadIdx.x < st) sm[threadIdx.x] += sm[threadIdx.x + st];
    __syncthreads();
  }
  if (threadIdx.x == 0) {
    float ov = sm[0] + 2.f * EPS + (float)NN * EPS * EPS;
    float t = fminf(fmaxf(temp_ptr[0], 0.1f), 5.0f);
    float mask = 1.f / (1.f + __expf(ov * t));   // sigmoid(-ov*t)
    rowscale[bh * NN + row] = mask / (mask + EPS);
  }
}

// ---- attn @ V (scaled) -> proj input [B,N,C] f16 --------------------------
// grid (N/256, BH, 2). 8 waves stacked along m; wave tile 32 rows x 64 cols.
__global__ __launch_bounds__(256) void attnv_kernel(
    const h16* __restrict__ attn0, const h16* __restrict__ attn1,
    const h16* __restrict__ Vt0, const h16* __restrict__ Vt1,
    const float* __restrict__ rowscale,
    h16* __restrict__ P0, h16* __restrict__ P1) {
  const int br = blockIdx.z;
  const long bh = blockIdx.y;
  const int b = (int)(bh / HH), h = (int)(bh % HH);
  const h16* Ab = (br ? attn1 : attn0) + bh * (long)NN * NN;
  const h16* Vb = (br ? Vt1 : Vt0) + bh * DD * NN;
  h16* P = br ? P1 : P0;
  const int wave = threadIdx.x >> 5;
  const int row0 = blockIdx.x * 256 + wave * 32;
  const int l = threadIdx.x & 31, cl = l & 15, lh = l >> 4;
  v8f acc[2][4] = {};
  for (int k0 = 0; k0 < NN; k0 += 32) {
    if (k0 + 64 < NN) {  // prefetch the A stream two chunks (64 k) ahead
      __builtin_prefetch(Ab + (long)(row0 + lh * 16 + (l & 15)) * NN + k0 + 64,
                         0, 1);
    }
    v16h a0 = frag_a(Ab, NN, row0, k0);
    v16h a1 = frag_a(Ab, NN, row0 + 16, k0);
#pragma unroll
    for (int j = 0; j < 4; ++j) {
      v16h bf = frag_b(Vb, NN, j * 16, k0);
      acc[0][j] = wmma32(a0, bf, acc[0][j]);
      acc[1][j] = wmma32(a1, bf, acc[1][j]);
    }
  }
#pragma unroll
  for (int mi = 0; mi < 2; ++mi)
#pragma unroll
    for (int v = 0; v < 8; ++v) {
      int n = row0 + mi * 16 + lh * 8 + v;
      float sc = rowscale[bh * NN + n];
#pragma unroll
      for (int j = 0; j < 4; ++j) {
        int d = j * 16 + cl;
        P[((long)b * NN + n) * CC + h * DD + d] = (h16)(acc[mi][j][v] * sc);
      }
    }
}

// ---- proj GEMM + bias -> d_out (f32); B chunks TDM-staged -----------------
__global__ __launch_bounds__(256) void proj_kernel(
    const h16* __restrict__ P0, const h16* __restrict__ P1,
    const h16* __restrict__ wt0, const h16* __restrict__ wt1,
    const float* __restrict__ bias0, const float* __restrict__ bias1,
    float* __restrict__ out) {
  __shared__ h16 wbuf[2][128 * WROW];
  const int br = blockIdx.z;
  const h16* A = br ? P1 : P0;
  const h16* wt = br ? wt1 : wt0;
  const float* bias = br ? bias1 : bias0;
  float* O = out + (long)br * MTOT * CC;
  const int wave = threadIdx.x >> 5;
  const int row0 = blockIdx.x * 128 + (wave >> 1) * 32;
  const int colB = blockIdx.y * 128;
  const int colL = (wave & 1) * 64;
  const h16* wslab = wt + (long)colB * CC;

  if (wave == 0) tdm_load_wchunk(wslab, lds_offset_of(&wbuf[0][0]));

  v8f acc[2][4] = {};
  for (int kc = 0; kc < CC / 32; ++kc) {
    const int k0 = kc * 32;
    __builtin_amdgcn_s_wait_tensorcnt((short)0);
    __syncthreads();
    if (wave == 0 && kc + 1 < CC / 32)
      tdm_load_wchunk(wslab + (kc + 1) * 32,
                      lds_offset_of(&wbuf[(kc + 1) & 1][0]));
    const h16* wb = &wbuf[kc & 1][0];
    v16h a0 = frag_a(A, CC, row0, k0);
    v16h a1 = frag_a(A, CC, row0 + 16, k0);
#pragma unroll
    for (int j = 0; j < 4; ++j) {
      v16h b = frag_b_lds(wb, colL + j * 16);
      acc[0][j] = wmma32(a0, b, acc[0][j]);
      acc[1][j] = wmma32(a1, b, acc[1][j]);
    }
  }
  const int l = threadIdx.x & 31, cl = l & 15, lh = l >> 4;
#pragma unroll
  for (int mi = 0; mi < 2; ++mi)
#pragma unroll
    for (int j = 0; j < 4; ++j)
#pragma unroll
      for (int v = 0; v < 8; ++v) {
        long row = row0 + mi * 16 + lh * 8 + v;
        int col = colB + colL + j * 16 + cl;
        O[row * CC + col] = acc[mi][j][v] + bias[col];
      }
}

// ---- in-place LayerNorm over C --------------------------------------------
__global__ __launch_bounds__(256) void ln_kernel(
    float* __restrict__ out,
    const float* __restrict__ g0, const float* __restrict__ b0,
    const float* __restrict__ g1, const float* __restrict__ b1) {
  const int br = blockIdx.y;
  float* row = out + (long)br * MTOT * CC + (long)blockIdx.x * CC;
  const float* g = br ? g1 : g0;
  const float* bb = br ? b1 : b0;
  __shared__ float sm[256];
  float x[3];
  float s = 0.f;
#pragma unroll
  for (int i = 0; i < 3; ++i) {
    x[i] = row[threadIdx.x + i * 256];
    s += x[i];
  }
  sm[threadIdx.x] = s;
  __syncthreads();
  for (int st = 128; st > 0; st >>= 1) {
    if (threadIdx.x < st) sm[threadIdx.x] += sm[threadIdx.x + st];
    __syncthreads();
  }
  float mu = sm[0] * (1.f / CC);
  __syncthreads();
  float vs = 0.f;
#pragma unroll
  for (int i = 0; i < 3; ++i) {
    float d = x[i] - mu;
    vs += d * d;
  }
  sm[threadIdx.x] = vs;
  __syncthreads();
  for (int st = 128; st > 0; st >>= 1) {
    if (threadIdx.x < st) sm[threadIdx.x] += sm[threadIdx.x + st];
    __syncthreads();
  }
  float r = rsqrtf(sm[0] * (1.f / CC) + 1e-5f);
#pragma unroll
  for (int i = 0; i < 3; ++i) {
    int c = threadIdx.x + i * 256;
    row[c] = (x[i] - mu) * r * g[c] + bb[c];
  }
}

// ---------------------------------------------------------------------------
extern "C" void kernel_launch(void* const* d_in, const int* in_sizes, int n_in,
                              void* d_out, int out_size, void* d_ws,
                              size_t ws_size, hipStream_t stream) {
  const float* x        = (const float*)d_in[0];
  const float* w_qkv_id = (const float*)d_in[1];
  const float* w_qkv_at = (const float*)d_in[2];
  const float* w_pr_id  = (const float*)d_in[3];
  const float* b_pr_id  = (const float*)d_in[4];
  const float* w_pr_at  = (const float*)d_in[5];
  const float* b_pr_at  = (const float*)d_in[6];
  const float* otemp    = (const float*)d_in[7];
  const float* ln_id_g  = (const float*)d_in[8];
  const float* ln_id_b  = (const float*)d_in[9];
  const float* ln_at_g  = (const float*)d_in[10];
  const float* ln_at_b  = (const float*)d_in[11];
  float* out = (float*)d_out;

  // ---- workspace carve-up (f16 elements; ~526 MB total) ----
  h16* ws = (h16*)d_ws;
  size_t o = 0;
  h16* xh   = ws + o; o += (size_t)MTOT * CC;          // x as f16
  h16* wq0  = ws + o; o += (size_t)CC * 3 * CC;        // transposed [2304,768]
  h16* wq1  = ws + o; o += (size_t)CC * 3 * CC;
  h16* wp0  = ws + o; o += (size_t)CC * CC;            // transposed [768,768]
  h16* wp1  = ws + o; o += (size_t)CC * CC;
  h16* Q0   = ws + o; o += (size_t)BH * NN * DD;
  h16* K0   = ws + o; o += (size_t)BH * NN * DD;
  h16* Vt0  = ws + o; o += (size_t)BH * NN * DD;
  h16* Q1   = ws + o; o += (size_t)BH * NN * DD;
  h16* K1   = ws + o; o += (size_t)BH * NN * DD;
  h16* Vt1  = ws + o; o += (size_t)BH * NN * DD;
  h16* P0   = ws + o; o += (size_t)MTOT * CC;          // proj inputs
  h16* P1   = ws + o; o += (size_t)MTOT * CC;
  h16* At0  = ws + o; o += (size_t)BH * NN * NN;       // f16 attention maps
  h16* At1  = ws + o; o += (size_t)BH * NN * NN;
  float* rowscale = (float*)(ws + o);                  // [BH*NN] f32

  // 1) converts
  cvt_f16_kernel<<<dim3((MTOT * CC + 255) / 256), 256, 0, stream>>>(x, xh, MTOT * CC);
  cvt_t_kernel<<<dim3((CC * 3 * CC + 255) / 256), 256, 0, stream>>>(w_qkv_id, wq0, CC, 3 * CC);
  cvt_t_kernel<<<dim3((CC * 3 * CC + 255) / 256), 256, 0, stream>>>(w_qkv_at, wq1, CC, 3 * CC);
  cvt_t_kernel<<<dim3((CC * CC + 255) / 256), 256, 0, stream>>>(w_pr_id, wp0, CC, CC);
  cvt_t_kernel<<<dim3((CC * CC + 255) / 256), 256, 0, stream>>>(w_pr_at, wp1, CC, CC);

  // 2) QKV GEMMs (both branches) with TDM-staged weights
  qkv_kernel<<<dim3(MTOT / 128, (3 * CC) / 128, 2), 256, 0, stream>>>(
      xh, wq0, wq1, Q0, K0, Vt0, Q1, K1, Vt1);

  // 3) scores + fused row softmax -> f16 attention maps
  scores_kernel<<<dim3(NN / 32, BH, 2), 256, 0, stream>>>(Q0, K0, Q1, K1, At0, At1);

  // 4) row overlap -> shared per-row scale
  overlap_kernel<<<dim3(NN, BH), 256, 0, stream>>>(At0, At1, otemp, rowscale);

  // 5) attn @ V with row-scale epilogue -> proj input [B,N,C] f16
  attnv_kernel<<<dim3(NN / 256, BH, 2), 256, 0, stream>>>(
      At0, At1, Vt0, Vt1, rowscale, P0, P1);

  // 6) proj GEMM (TDM-staged weights) + bias -> d_out, then in-place LN
  proj_kernel<<<dim3(MTOT / 128, CC / 128, 2), 256, 0, stream>>>(
      P0, P1, wp0, wp1, b_pr_id, b_pr_at, out);
  ln_kernel<<<dim3(MTOT, 2), 256, 0, stream>>>(out, ln_id_g, ln_id_b, ln_at_g, ln_at_b);
}
